// AGRNNModel_47691316855473
// MI455X (gfx1250) — matrix operand order repaired and checked by reading, not compiled, not run on hardware
//
#include <hip/hip_runtime.h>
#include <math.h>
#include <stdint.h>

// ---- problem constants (match reference) ----
#define BB 32
#define TT 12
#define FF 2
#define LL 1024
#define HH 64
#define HOR 12

typedef __attribute__((ext_vector_type(16))) _Float16 v16h;
typedef __attribute__((ext_vector_type(8)))  _Float16 v8h;
typedef __attribute__((ext_vector_type(4)))  _Float16 v4h;
typedef __attribute__((ext_vector_type(8)))  float    v8f;

// ---------------- WMMA helpers (CDNA5 gfx1250, wave32) ----------------
__device__ __forceinline__ v8f wmma_f16(v16h a, v16h b, v8f c) {
  return __builtin_amdgcn_wmma_f32_16x16x32_f16(false, a, false, b, (short)0, c,
                                                false, false);
}

// A fragment: M x K = 16x32, lane m = lane&15, g = lane>>4.
__device__ __forceinline__ v16h frag_a(const _Float16* row, int g) {
  v8h lo = *(const v8h*)(row + g * 8);
  v8h hi = *(const v8h*)(row + 16 + g * 8);
  return __builtin_shufflevector(lo, hi, 0, 1, 2, 3, 4, 5, 6, 7, 8, 9, 10, 11,
                                 12, 13, 14, 15);
}

// B fragment: K x N = 32x16, lane n = lane&15, g = lane>>4.
__device__ __forceinline__ v16h frag_b(const _Float16* row, int g) {
  v8h lo = *(const v8h*)(row + g * 8);
  v8h hi = *(const v8h*)(row + 16 + g * 8);
  return __builtin_shufflevector(lo, hi, 0, 1, 2, 3, 4, 5, 6, 7, 8, 9, 10, 11,
                                 12, 13, 14, 15);
}

__device__ __forceinline__ v4h cvt4(float4 v) {
  v4h h;
  h[0] = (_Float16)v.x; h[1] = (_Float16)v.y;
  h[2] = (_Float16)v.z; h[3] = (_Float16)v.w;
  return h;
}

__device__ __forceinline__ float sigmoidf_(float x) {
  return 1.0f / (1.0f + expf(-x));
}

// ---------------- zero (float4 granularity) ----------------
__global__ __launch_bounds__(256) void k_zero(float4* p, int n4) {
  int i = blockIdx.x * 256 + threadIdx.x;
  if (i < n4) p[i] = make_float4(0.f, 0.f, 0.f, 0.f);
}

// ---------------- f32 -> f16 bulk convert (weights, once per launch) --------
__global__ __launch_bounds__(256) void k_cvt_f16(const float4* __restrict__ src,
                                                 v4h* __restrict__ dst, int n4) {
  int i = blockIdx.x * 256 + threadIdx.x;
  if (i < n4) dst[i] = cvt4(src[i]);
}

// ---------------- pack x[:, t, :] -> inrH f16 [(b*F+f), l] ----------------
__global__ __launch_bounds__(256) void k_pack_x(const float4* __restrict__ x,
                                                v4h* __restrict__ inrH, int t) {
  int idx = blockIdx.x * 256 + threadIdx.x;  // quads
  if (idx >= BB * FF * LL / 4) return;
  int b = idx / (FF * LL / 4);
  int rem = idx - b * (FF * LL / 4);
  inrH[b * (FF * LL / 4) + rem] =
      cvt4(x[(size_t)b * TT * (FF * LL / 4) + (size_t)t * (FF * LL / 4) + rem]);
}

// ---------------- attn GEMM (f16 in, f32 out) ----------------
// outT[n][m] = sum_k A[m][k] * W[n][k].  A: [M x 1024] f16, W: [1024 x 1024] f16.
// Block tile 64x64, K-chunk 64, LDS ping-pong fed by async global->LDS copies,
// manually software-pipelined with compile-time buffer indices.
#define KC 64
#define LDP 72  // padded LDS row stride (halves); 144B, 16B aligned

__global__ __launch_bounds__(256) void k_attn_gemm(
    const _Float16* __restrict__ A, const _Float16* __restrict__ W,
    float* __restrict__ outT, int M) {
  const int tid = threadIdx.x;
  const int lane = tid & 31, wave = tid >> 5;
  const int g = lane >> 4, ln = lane & 15;
  const int n_blk = blockIdx.x * 64;
  const int m_blk = blockIdx.y * 64;
  const int m_t = wave & 3;   // 0..3  -> 16 rows each
  const int n_t = wave >> 2;  // 0..1  -> 32 cols each

  __shared__ __align__(16) _Float16 As[2][64 * LDP];
  __shared__ __align__(16) _Float16 Bs[2][64 * LDP];

  // staging: 64 rows x 64 halves = 512 x 16B transfers per matrix,
  // 2 per thread per matrix.  q = tid + i*256 : r = q>>3, c8 = q&7.
  const int r0 = tid >> 3, c0 = tid & 7;
  const int r1 = (tid + 256) >> 3, c1 = (tid + 256) & 7;

  // LDS byte offsets (aperture model: low 32 bits of generic ptr = LDS offset)
  const unsigned laBase[2] = {(unsigned)(uintptr_t)&As[0][0],
                              (unsigned)(uintptr_t)&As[1][0]};
  const unsigned lbBase[2] = {(unsigned)(uintptr_t)&Bs[0][0],
                              (unsigned)(uintptr_t)&Bs[1][0]};
  const unsigned o0 = (unsigned)(r0 * LDP + c0 * 8) * 2;
  const unsigned o1 = (unsigned)(r1 * LDP + c1 * 8) * 2;

#define ASYNC_CP(ldsaddr, gaddr)                                              \
  asm volatile("global_load_async_to_lds_b128 %0, %1, off" ::"v"(ldsaddr),    \
               "v"(gaddr)                                                     \
               : "memory")

#define ISSUE(bf, kk)                                                         \
  {                                                                           \
    unsigned long long gA0 = (unsigned long long)(uintptr_t)(                 \
        A + (size_t)(m_blk + r0) * 1024 + (kk) + c0 * 8);                     \
    unsigned long long gA1 = (unsigned long long)(uintptr_t)(                 \
        A + (size_t)(m_blk + r1) * 1024 + (kk) + c1 * 8);                     \
    unsigned long long gB0 = (unsigned long long)(uintptr_t)(                 \
        W + (size_t)(n_blk + r0) * 1024 + (kk) + c0 * 8);                     \
    unsigned long long gB1 = (unsigned long long)(uintptr_t)(                 \
        W + (size_t)(n_blk + r1) * 1024 + (kk) + c1 * 8);                     \
    ASYNC_CP(laBase[bf] + o0, gA0);                                           \
    ASYNC_CP(laBase[bf] + o1, gA1);                                           \
    ASYNC_CP(lbBase[bf] + o0, gB0);                                           \
    ASYNC_CP(lbBase[bf] + o1, gB1);                                           \
  }

#define WAIT_ASYNC() asm volatile("s_wait_asynccnt 0x0" ::: "memory")

  v8f c0acc = {}; v8f c1acc = {};
  const int a_row = (m_t * 16 + ln) * LDP;
  const int b_row0 = (n_t * 32 + ln) * LDP;
  const int b_row1 = (n_t * 32 + 16 + ln) * LDP;

#define COMPUTE(bf)                                                           \
  {                                                                           \
    v16h a0 = frag_a(&As[bf][a_row], g);                                      \
    v16h a1 = frag_a(&As[bf][a_row + 32], g);                                 \
    c0acc = wmma_f16(a0, frag_b(&Bs[bf][b_row0], g), c0acc);                  \
    c0acc = wmma_f16(a1, frag_b(&Bs[bf][b_row0 + 32], g), c0acc);             \
    c1acc = wmma_f16(a0, frag_b(&Bs[bf][b_row1], g), c1acc);                  \
    c1acc = wmma_f16(a1, frag_b(&Bs[bf][b_row1 + 32], g), c1acc);             \
  }

  // 16 K-chunks total.  Prologue: chunk 0 -> buf0.
  ISSUE(0, 0);
  WAIT_ASYNC();
  __syncthreads();
  // Steady state, unrolled x2 so buffer indices are compile-time constants:
  // 7 iterations stage chunks 1..14 and compute chunks 0..13.
  for (int kk = KC; kk < 15 * KC; kk += 2 * KC) {
    ISSUE(1, kk);           // stage odd chunk under WMMAs on buf0
    COMPUTE(0);
    WAIT_ASYNC();
    __syncthreads();
    ISSUE(0, kk + KC);      // stage even chunk under WMMAs on buf1
    COMPUTE(1);
    WAIT_ASYNC();
    __syncthreads();
  }
  // Peel: stage chunk 15 -> buf1, compute chunk 14 (buf0), then chunk 15.
  ISSUE(1, 15 * KC);
  COMPUTE(0);
  WAIT_ASYNC();
  __syncthreads();
  COMPUTE(1);

#pragma unroll
  for (int j = 0; j < 8; ++j) {
    int m = m_blk + m_t * 16 + g * 8 + j;
    int n0 = n_blk + n_t * 32 + ln;
    outT[(size_t)n0 * M + m] = c0acc[j];
    outT[(size_t)(n0 + 16) * M + m] = c1acc[j];
  }
}

// ---------------- per-link GRU gates ----------------
// h2T: [L][B*H] f32 attention'd hidden; in2T: [L][B*F] f32 attention'd input
// writes hOut f32 [(b*H+k)][l] and hOutH f16 (same layout, feeds next GEMM)
__global__ __launch_bounds__(256) void k_gates(
    const float* __restrict__ h2T, const float* __restrict__ in2T,
    const float* __restrict__ Wrh, const float* __restrict__ brh,
    const float* __restrict__ Wri, const float* __restrict__ bri,
    const float* __restrict__ Wzh, const float* __restrict__ bzh,
    const float* __restrict__ Wzi, const float* __restrict__ bzi,
    const float* __restrict__ Wnh, const float* __restrict__ bnh,
    const float* __restrict__ Wni, const float* __restrict__ bni,
    float* __restrict__ hOut, _Float16* __restrict__ hOutH) {
  const int l = blockIdx.x;
  const int tid = threadIdx.x;
  const int lane = tid & 31, wave = tid >> 5;
  const int g = lane >> 4, ln = lane & 15;

  __shared__ __align__(16) _Float16 Ah[32 * LDP];     // [b][h] f16, padded
  __shared__ __align__(16) _Float16 Bw[3][64 * LDP];  // [k][h] f16 (transposed)
  __shared__ float Sin[BB * FF];

  const float* h2l = h2T + (size_t)l * (BB * HH);
  const float* Wh0 = Wrh + (size_t)l * (HH * HH);
  const float* Wh1 = Wzh + (size_t)l * (HH * HH);
  const float* Wh2 = Wnh + (size_t)l * (HH * HH);

#pragma unroll
  for (int i = 0; i < 2; ++i) {
    int q = tid + i * 256;
    int c4 = q & 15;
    float4 av = *(const float4*)&h2l[q * 4];
    *(v4h*)&Ah[(q >> 4) * LDP + c4 * 4] = cvt4(av);
  }
#pragma unroll
  for (int i = 0; i < 4; ++i) {
    int q = tid + i * 256;
    int hh = q >> 4, k4 = q & 15;
    float4 w0 = *(const float4*)&Wh0[hh * 64 + k4 * 4];
    float4 w1 = *(const float4*)&Wh1[hh * 64 + k4 * 4];
    float4 w2 = *(const float4*)&Wh2[hh * 64 + k4 * 4];
    v4h h0 = cvt4(w0), h1 = cvt4(w1), h2 = cvt4(w2);
#pragma unroll
    for (int j = 0; j < 4; ++j) {
      Bw[0][(k4 * 4 + j) * LDP + hh] = h0[j];
      Bw[1][(k4 * 4 + j) * LDP + hh] = h1[j];
      Bw[2][(k4 * 4 + j) * LDP + hh] = h2[j];
    }
  }
  if (tid < BB * FF) Sin[tid] = in2T[(size_t)l * (BB * FF) + tid];
  __syncthreads();

  const int m_t = wave & 1;   // 0..1 (batch tiles of 16)
  const int n_t = wave >> 1;  // 0..3 (output-k tiles of 16)

  v16h a0 = frag_a(&Ah[(m_t * 16 + ln) * LDP], g);
  v16h a1 = frag_a(&Ah[(m_t * 16 + ln) * LDP + 32], g);

  const _Float16* brow0 = &Bw[0][(n_t * 16 + ln) * LDP];
  const _Float16* brow1 = &Bw[1][(n_t * 16 + ln) * LDP];
  const _Float16* brow2 = &Bw[2][(n_t * 16 + ln) * LDP];

  v8f accR = {}, accZ = {}, accN = {};
  accR = wmma_f16(a0, frag_b(brow0, g), accR);
  accR = wmma_f16(a1, frag_b(brow0 + 32, g), accR);
  accZ = wmma_f16(a0, frag_b(brow1, g), accZ);
  accZ = wmma_f16(a1, frag_b(brow1 + 32, g), accZ);
  accN = wmma_f16(a0, frag_b(brow2, g), accN);
  accN = wmma_f16(a1, frag_b(brow2 + 32, g), accN);

  const int k = n_t * 16 + ln;
  const float b_rh = brh[(size_t)k * LL + l], b_ri = bri[(size_t)k * LL + l];
  const float b_zh = bzh[(size_t)k * LL + l], b_zi = bzi[(size_t)k * LL + l];
  const float b_nh = bnh[(size_t)k * LL + l], b_ni = bni[(size_t)k * LL + l];
  const float wri0 = Wri[(size_t)l * (FF * HH) + k];
  const float wri1 = Wri[(size_t)l * (FF * HH) + HH + k];
  const float wzi0 = Wzi[(size_t)l * (FF * HH) + k];
  const float wzi1 = Wzi[(size_t)l * (FF * HH) + HH + k];
  const float wni0 = Wni[(size_t)l * (FF * HH) + k];
  const float wni1 = Wni[(size_t)l * (FF * HH) + HH + k];

#pragma unroll
  for (int j = 0; j < 8; ++j) {
    int b = m_t * 16 + g * 8 + j;
    float x0 = Sin[b * 2 + 0], x1 = Sin[b * 2 + 1];
    float r = sigmoidf_(accR[j] + b_rh + x0 * wri0 + x1 * wri1 + b_ri);
    float z = sigmoidf_(accZ[j] + b_zh + x0 * wzi0 + x1 * wzi1 + b_zi);
    float n = tanhf(r * (accN[j] + b_nh) + x0 * wni0 + x1 * wni1 + b_ni);
    float hv = h2l[b * HH + k];
    float outv = (1.0f - z) * n + z * hv;
    hOut[(size_t)(b * HH + k) * LL + l] = outv;
    hOutH[(size_t)(b * HH + k) * LL + l] = (_Float16)outv;
  }
}

// ---------------- decoder head ----------------
__global__ __launch_bounds__(256) void k_dec(const float* __restrict__ h,
                                             const float* __restrict__ fcw,
                                             const float* __restrict__ fcb,
                                             float* __restrict__ out,
                                             _Float16* __restrict__ inrH,
                                             int step) {
  int idx = blockIdx.x * 256 + threadIdx.x;  // B*L
  if (idx >= BB * LL) return;
  int b = idx >> 10, l = idx & 1023;
  float s = fcb[l];
  const float* hp = h + (size_t)b * HH * LL + l;
  const float* wp = fcw + (size_t)l * HH;
#pragma unroll
  for (int hh = 0; hh < HH; ++hh) s += hp[(size_t)hh * LL] * wp[hh];
  out[(size_t)b * HOR * LL + (size_t)step * LL + l] = s;
  inrH[(size_t)(b * FF + 0) * LL + l] = (_Float16)s;
  inrH[(size_t)(b * FF + 1) * LL + l] = (_Float16)0.0f;
}

// ---------------- host orchestration ----------------
extern "C" void kernel_launch(void* const* d_in, const int* in_sizes, int n_in,
                              void* d_out, int out_size, void* d_ws,
                              size_t ws_size, hipStream_t stream) {
  (void)in_sizes; (void)n_in; (void)out_size; (void)ws_size;
  const float* x   = (const float*)d_in[0];
  const float* Wia = (const float*)d_in[1];
  const float* Wha = (const float*)d_in[2];
  const float* Wrh = (const float*)d_in[3];  const float* brh = (const float*)d_in[4];
  const float* Wri = (const float*)d_in[5];  const float* bri = (const float*)d_in[6];
  const float* Wzh = (const float*)d_in[7];  const float* bzh = (const float*)d_in[8];
  const float* Wzi = (const float*)d_in[9];  const float* bzi = (const float*)d_in[10];
  const float* Wnh = (const float*)d_in[11]; const float* bnh = (const float*)d_in[12];
  const float* Wni = (const float*)d_in[13]; const float* bni = (const float*)d_in[14];
  const float* fcw = (const float*)d_in[15]; const float* fcb = (const float*)d_in[16];
  float* out = (float*)d_out;

  char* ws = (char*)d_ws;
  const size_t MB = 1024 * 1024;
  float*     h    = (float*)(ws);                 // [B*H][L] f32   8 MB
  float*     h2T  = (float*)(ws + 8 * MB);        // [L][B*H] f32   8 MB
  _Float16*  hH   = (_Float16*)(ws + 16 * MB);    // [B*H][L] f16   4 MB
  _Float16*  WiaH = (_Float16*)(ws + 20 * MB);    // [L][L]   f16   2 MB
  _Float16*  WhaH = (_Float16*)(ws + 22 * MB);    // [L][L]   f16   2 MB
  _Float16*  inrH = (_Float16*)(ws + 24 * MB);    // [B*F][L] f16   128 KB
  float*     in2T = (float*)(ws + 24 * MB + 256 * 1024);  // [L][B*F] f32 256 KB

  dim3 blk(256);

  // one-time per launch: weight conversion + state zero-init
  const int nW4 = LL * LL / 4;  // 256K quads
  k_cvt_f16<<<(nW4 + 255) / 256, blk, 0, stream>>>((const float4*)Wia,
                                                   (v4h*)WiaH, nW4);
  k_cvt_f16<<<(nW4 + 255) / 256, blk, 0, stream>>>((const float4*)Wha,
                                                   (v4h*)WhaH, nW4);
  const int nH4 = BB * HH * LL / 4;
  k_zero<<<(nH4 + 255) / 256, blk, 0, stream>>>((float4*)h, nH4);
  const int nHh4 = BB * HH * LL / 8;  // f16 buffer in float4 units
  k_zero<<<(nHh4 + 255) / 256, blk, 0, stream>>>((float4*)hH, nHh4);

  dim3 gIn(LL / 64, (BB * FF) / 64);   // 16 x 1
  dim3 gHid(LL / 64, (BB * HH) / 64);  // 16 x 32

  // ---- encoder ----
  for (int t = 0; t < TT; ++t) {
    k_pack_x<<<(BB * FF * LL / 4 + 255) / 256, blk, 0, stream>>>(
        (const float4*)x, (v4h*)inrH, t);
    k_attn_gemm<<<gIn, blk, 0, stream>>>(inrH, WiaH, in2T, BB * FF);
    k_attn_gemm<<<gHid, blk, 0, stream>>>(hH, WhaH, h2T, BB * HH);
    k_gates<<<LL, blk, 0, stream>>>(h2T, in2T, Wrh, brh, Wri, bri, Wzh, bzh,
                                    Wzi, bzi, Wnh, bnh, Wni, bni, h, hH);
  }

  // ---- decoder ----
  for (int s = 0; s < HOR; ++s) {
    k_dec<<<(BB * LL + 255) / 256, blk, 0, stream>>>(h, fcw, fcb, out, inrH, s);
    k_attn_gemm<<<gIn, blk, 0, stream>>>(inrH, WiaH, in2T, BB * FF);
    k_attn_gemm<<<gHid, blk, 0, stream>>>(hH, WhaH, h2T, BB * HH);
    k_gates<<<LL, blk, 0, stream>>>(h2T, in2T, Wrh, brh, Wri, bri, Wzh, bzh,
                                    Wzi, bzi, Wnh, bnh, Wni, bni, h, hH);
  }
}
